// CrossmodalSemanticsCalibration_56255481643639
// MI455X (gfx1250) — compile-verified, matrix operand order checked by reading, not run on hardware
//
#include <hip/hip_runtime.h>
#include <hip/hip_bf16.h>
#include <math.h>

typedef __bf16 bf16;
typedef __bf16 v16bf __attribute__((ext_vector_type(16)));
typedef float  v8f   __attribute__((ext_vector_type(8)));

#define N_B   4
#define L_TOK 32768
#define S_TOK 32768
#define D_F   96
#define D2_F  192
#define H_N   8
#define HD_N  12
#define NL    (N_B * L_TOK)
#define NS    (N_B * S_TOK)

#define ZERO8 ((v8f){0.f,0.f,0.f,0.f,0.f,0.f,0.f,0.f})
#define WFENCE() __builtin_amdgcn_wave_barrier()

// ---- CDNA5 async global->LDS copy (ASYNCcnt path, GV addressing mode) -----
__device__ __forceinline__ void async_ld_b128(unsigned int lds_off,
                                              unsigned long long gaddr) {
  asm volatile("global_load_async_to_lds_b128 %0, %1, off"
               :
               : "v"(lds_off), "v"(gaddr)
               : "memory");
}
__device__ __forceinline__ void wait_asynccnt0() {
  asm volatile("s_wait_asynccnt 0x0" ::: "memory");
}

// ---- WMMA fragment loader -------------------------------------------------
// 16-bit 16x32 A layout (ISA 7.12.2): lane<16 -> row, K in {0..7,16..23};
// lane>=16 -> row, K in {8..15,24..31}; pairs packed per VGPR.
// B is stored pre-transposed ([N][K] row-major) so the same pattern applies
// with rsel = output column.
__device__ __forceinline__ v16bf load_frag(const bf16* __restrict__ P, int ld,
                                           int rsel, int g) {
  v16bf f;
#pragma unroll
  for (int j = 0; j < 8; ++j) {
    const int kb = ((j & 4) << 2) + (g << 3) + ((j & 3) << 1);
    f[2 * j]     = P[rsel * ld + kb];
    f[2 * j + 1] = P[rsel * ld + kb + 1];
  }
  return f;
}

// C = A(16 x 32*KT, LDS bf16, stride lda) * B^T(Bt: [16*NT][32*KT] bf16, stride ldb)
template <int KT, int NT>
__device__ __forceinline__ void gemm16(const bf16* __restrict__ A, int lda,
                                       const bf16* __restrict__ Bt, int ldb,
                                       v8f acc[NT]) {
  const int lane = (int)(threadIdx.x & 31u);
  const int rsel = lane & 15, g = lane >> 4;
#pragma unroll
  for (int kt = 0; kt < KT; ++kt) {
    const v16bf a = load_frag(A + kt * 32, lda, rsel, g);
#pragma unroll
    for (int nt = 0; nt < NT; ++nt) {
      const v16bf b = load_frag(Bt + nt * 16 * ldb + kt * 32, ldb, rsel, g);
      acc[nt] = __builtin_amdgcn_wmma_f32_16x16x32_bf16(
          false, a, false, b, (short)0, acc[nt], false, false);
    }
  }
}
// C layout (ISA): VGPR r, lanes 0-15 -> M=r, lanes 16-31 -> M=8+r; N = lane&15.

// ---- k0: weight convert+transpose f32 -> bf16 [N][K] ----------------------
__global__ void k0_convT(const float* __restrict__ src, bf16* __restrict__ dst,
                         int K, int Ncols) {
  const int idx = blockIdx.x * 256 + threadIdx.x;
  if (idx < K * Ncols) {
    const int k = idx / Ncols, n = idx % Ncols;
    dst[n * K + k] = (bf16)src[idx];
  }
}

__global__ void k0_zero(float* __restrict__ p, int n) {
  const int i = blockIdx.x * 256 + threadIdx.x;
  if (i < n) p[i] = 0.f;
}

// ---- k1: LN -> shortcut -> x1 -> q (per 16-row tile, one wave) ------------
__global__ __launch_bounds__(256) void k1_qkv(
    const float* __restrict__ mr, const float* __restrict__ ln1g,
    const float* __restrict__ ln1b, const bf16* __restrict__ wqkvT,
    const bf16* __restrict__ wqkv2T, const bf16* __restrict__ cwqT,
    float* __restrict__ shortcut, float* __restrict__ x1,
    float* __restrict__ qout) {
  __shared__ bf16 hb[8][16][D_F];
  const int w = (int)(threadIdx.x >> 5u), lane = (int)(threadIdx.x & 31u);
  const int tile = blockIdx.x * 8 + w;
  const long base = (long)tile * 16;
  bf16(*hw)[D_F] = hb[w];
  const int rsel = lane & 15, g = lane >> 4;

  if (lane < 16) {  // LayerNorm one row per lane
    const float* row = mr + (base + lane) * D_F;
    float s = 0.f, s2 = 0.f;
    for (int c = 0; c < D_F; ++c) { const float v = row[c]; s += v; s2 += v * v; }
    const float m = s * (1.f / D_F);
    const float rv = rsqrtf(s2 * (1.f / D_F) - m * m + 1e-5f);
    for (int c = 0; c < D_F; ++c)
      hw[lane][c] = (bf16)(((row[c] - m) * rv) * ln1g[c] + ln1b[c]);
  }
  WFENCE();

  v8f acc[6];
#pragma unroll
  for (int i = 0; i < 6; ++i) acc[i] = ZERO8;
  gemm16<3, 6>(&hw[0][0], D_F, wqkvT, D_F, acc);  // shortcut = x0 @ w_qkv
  WFENCE();
#pragma unroll
  for (int nt = 0; nt < 6; ++nt)
#pragma unroll
    for (int r = 0; r < 8; ++r) {
      const int m = r + 8 * g, col = nt * 16 + rsel;
      const float v = acc[nt][r];
      shortcut[(base + m) * D_F + col] = v;
      hw[m][col] = (bf16)v;
    }
  WFENCE();

#pragma unroll
  for (int i = 0; i < 6; ++i) acc[i] = ZERO8;
  gemm16<3, 6>(&hw[0][0], D_F, wqkv2T, D_F, acc);  // x1 = shortcut @ w_qkv2
  WFENCE();
#pragma unroll
  for (int nt = 0; nt < 6; ++nt)
#pragma unroll
    for (int r = 0; r < 8; ++r) {
      const int m = r + 8 * g, col = nt * 16 + rsel;
      const float v = acc[nt][r];
      x1[(base + m) * D_F + col] = v;
      hw[m][col] = (bf16)v;
    }
  WFENCE();

#pragma unroll
  for (int i = 0; i < 6; ++i) acc[i] = ZERO8;
  gemm16<3, 6>(&hw[0][0], D_F, cwqT, D_F, acc);  // q = x1 @ c_wq
#pragma unroll
  for (int nt = 0; nt < 6; ++nt)
#pragma unroll
    for (int r = 0; r < 8; ++r)
      qout[(base + r + 8 * g) * D_F + nt * 16 + rsel] = acc[nt][r];
}

// ---- k2: k/v projections + KV / Ksum reduction ----------------------------
// Each wave processes 8 consecutive 16-row tiles, accumulating KV (1152 =
// 36*32) and Ksum (96 = 3*32) partials in registers, then does ONE batch of
// atomics -> 8x less atomic contention on the 5KB accumulator.
#define K2_TPW 8
__global__ __launch_bounds__(128) void k2_kv(
    const float* __restrict__ warp, const bf16* __restrict__ cwkT,
    const bf16* __restrict__ cwvT, float* __restrict__ kvg,
    float* __restrict__ ksumg) {
  __shared__ bf16 hb[4][16][D_F];
  __shared__ float ktl[4][16][D_F];
  __shared__ float vtl[4][16][D_F];
  const int w = (int)(threadIdx.x >> 5u), lane = (int)(threadIdx.x & 31u);
  const int waveId = blockIdx.x * 4 + w;
  const int rsel = lane & 15, g = lane >> 4;

  float kvacc[36];
  float ksacc[3];
#pragma unroll
  for (int u = 0; u < 36; ++u) kvacc[u] = 0.f;
#pragma unroll
  for (int u = 0; u < 3; ++u) ksacc[u] = 0.f;

  for (int t = 0; t < K2_TPW; ++t) {
    const long base = ((long)waveId * K2_TPW + t) * 16;

    for (int idx = lane; idx < 16 * D_F; idx += 32) {
      const int r = idx / D_F, c = idx % D_F;
      hb[w][r][c] = (bf16)warp[(base + r) * D_F + c];
    }
    WFENCE();

    v8f acc[6];
#pragma unroll
    for (int i = 0; i < 6; ++i) acc[i] = ZERO8;
    gemm16<3, 6>(&hb[w][0][0], D_F, cwkT, D_F, acc);  // k
    WFENCE();
#pragma unroll
    for (int nt = 0; nt < 6; ++nt)
#pragma unroll
      for (int r = 0; r < 8; ++r) {
        const float v = acc[nt][r];  // K' = elu(k)+1
        ktl[w][r + 8 * g][nt * 16 + rsel] = v > 0.f ? v + 1.f : expf(v);
      }

#pragma unroll
    for (int i = 0; i < 6; ++i) acc[i] = ZERO8;
    gemm16<3, 6>(&hb[w][0][0], D_F, cwvT, D_F, acc);  // v
    WFENCE();
#pragma unroll
    for (int nt = 0; nt < 6; ++nt)
#pragma unroll
      for (int r = 0; r < 8; ++r)
        vtl[w][r + 8 * g][nt * 16 + rsel] = acc[nt][r];
    WFENCE();

    // KV[h][d][e] += sum_r K'[r][h*12+d] * V[r][h*12+e]  (v/S * S cancels)
#pragma unroll
    for (int u = 0; u < 36; ++u) {
      const int idx = lane + 32 * u;
      const int h = idx / 144, rem = idx % 144, d = rem / 12, e = rem % 12;
      const int ck = h * 12 + d, cv = h * 12 + e;
      float s = 0.f;
#pragma unroll
      for (int r = 0; r < 16; ++r) s += ktl[w][r][ck] * vtl[w][r][cv];
      kvacc[u] += s;
    }
#pragma unroll
    for (int u = 0; u < 3; ++u) {
      const int c = lane + 32 * u;
      float s = 0.f;
#pragma unroll
      for (int r = 0; r < 16; ++r) s += ktl[w][r][c];
      ksacc[u] += s;
    }
    WFENCE();
  }

  const int n = (int)(((long)waveId * K2_TPW * 16) / S_TOK);  // 128 rows, one n
#pragma unroll
  for (int u = 0; u < 36; ++u)
    atomicAdd(&kvg[n * 1152 + lane + 32 * u], kvacc[u]);
#pragma unroll
  for (int u = 0; u < 3; ++u)
    atomicAdd(&ksumg[n * D_F + lane + 32 * u], ksacc[u]);
}

// ---- k3: attention apply + full tail --------------------------------------
__global__ __launch_bounds__(64) void k3_final(
    const float* __restrict__ x1g, const float* __restrict__ scg,
    const float* __restrict__ qg, const float* __restrict__ kvg,
    const float* __restrict__ ksumg, const bf16* __restrict__ cwmergeT,
    const bf16* __restrict__ cwm1T, const bf16* __restrict__ cwm2T,
    const bf16* __restrict__ wfc1T, const bf16* __restrict__ wfc2T,
    const float* __restrict__ cln1g, const float* __restrict__ cln1b,
    const float* __restrict__ cln2g, const float* __restrict__ cln2b,
    const float* __restrict__ ln2g, const float* __restrict__ ln2b,
    const float* __restrict__ bfc1, const float* __restrict__ bfc2,
    float* __restrict__ out) {
  __shared__ float fA[2][16][D_F];   // x1 (f32, kept for residual)
  __shared__ float fB[2][16][D2_F];  // f32 scratch: Qt -> mm -> msg2 | [96:192]=shortcut (async DMA)
  __shared__ float zb[2][16][H_N];
  __shared__ bf16 hA[2][16][D2_F];   // bf16 A-staging (x1|msg_ln, then cat2)
  __shared__ bf16 hB[2][16][D2_F];   // bf16 A-staging (msg, t2, t3)
  const int w = (int)(threadIdx.x >> 5u), lane = (int)(threadIdx.x & 31u);
  const int tile = blockIdx.x * 2 + w;
  const long base = (long)tile * 16;
  const int n = (int)(base / L_TOK);
  const int rsel = lane & 15, g = lane >> 4;

  __builtin_prefetch(cwm1T, 0, 3);   // global_prefetch: warm big weights in L2
  __builtin_prefetch(wfc1T, 0, 3);

  // Kick off async DMA of the shortcut tile into fB[:,96:192] NOW; it is not
  // consumed until the concat-LN, so it overlaps attention + three GEMMs.
  {
    const unsigned int lds0 = (unsigned int)(unsigned long long)(&fB[w][0][0]);
    const unsigned long long gbase =
        (unsigned long long)(const void*)(scg + base * D_F);
#pragma unroll
    for (int u = 0; u < 12; ++u) {  // 384 b128 chunks / 32 lanes
      const int i = lane + 32 * u;
      const int r = i / 24, cc = (i % 24) * 4;
      async_ld_b128(lds0 + (unsigned int)((r * D2_F + D_F + cc) * 4),
                    gbase + (unsigned long long)((r * D_F + cc) * 4));
    }
  }

  // load x1 (f32 + bf16) and Q' = elu(q)+1
  for (int idx = lane; idx < 16 * D_F; idx += 32) {
    const int r = idx / D_F, c = idx % D_F;
    const float v = x1g[(base + r) * D_F + c];
    fA[w][r][c] = v;
    hA[w][r][c] = (bf16)v;
    const float q = qg[(base + r) * D_F + c];
    fB[w][r][c] = q > 0.f ? q + 1.f : expf(q);
  }
  WFENCE();

  // Z = 1 / (Q . Ksum + eps)
  const float* ks = ksumg + n * D_F;
  for (int idx = lane; idx < 16 * H_N; idx += 32) {
    const int r = idx >> 3, h = idx & 7;
    float s = 0.f;
#pragma unroll
    for (int d = 0; d < HD_N; ++d) s += fB[w][r][h * 12 + d] * ks[h * 12 + d];
    zb[w][r][h] = 1.f / (s + 1e-6f);
  }
  WFENCE();

  // msg = (Q . KV) * Z  -> hB[:,0:96] bf16
  const float* kv = kvg + n * 1152;
  for (int idx = lane; idx < 16 * D_F; idx += 32) {
    const int r = idx / D_F, c = idx % D_F, h = c / 12, e = c % 12;
    float s = 0.f;
#pragma unroll
    for (int d = 0; d < HD_N; ++d)
      s += fB[w][r][h * 12 + d] * kv[h * 144 + d * 12 + e];
    hB[w][r][c] = (bf16)(s * zb[w][r][h]);
  }
  WFENCE();

  v8f acc6[6];
  v8f acc12[12];

  // mm = msg @ c_wmerge -> fB[:,0:96] f32
#pragma unroll
  for (int i = 0; i < 6; ++i) acc6[i] = ZERO8;
  gemm16<3, 6>(&hB[w][0][0], D2_F, cwmergeT, D_F, acc6);
  WFENCE();
#pragma unroll
  for (int nt = 0; nt < 6; ++nt)
#pragma unroll
    for (int r = 0; r < 8; ++r)
      fB[w][r + 8 * g][nt * 16 + rsel] = acc6[nt][r];
  WFENCE();

  // LN(c_ln1) -> hA[:,96:192]
  if (lane < 16) {
    float s = 0.f, s2 = 0.f;
    for (int c = 0; c < D_F; ++c) { const float v = fB[w][lane][c]; s += v; s2 += v * v; }
    const float m = s * (1.f / D_F);
    const float rv = rsqrtf(s2 * (1.f / D_F) - m * m + 1e-5f);
    for (int c = 0; c < D_F; ++c)
      hA[w][lane][D_F + c] =
          (bf16)(((fB[w][lane][c] - m) * rv) * cln1g[c] + cln1b[c]);
  }
  WFENCE();

  // t2 = relu(cat(x1,msg_ln) @ c_wm1) -> hB bf16 [16][192]
#pragma unroll
  for (int i = 0; i < 12; ++i) acc12[i] = ZERO8;
  gemm16<6, 12>(&hA[w][0][0], D2_F, cwm1T, D2_F, acc12);
  WFENCE();
#pragma unroll
  for (int nt = 0; nt < 12; ++nt)
#pragma unroll
    for (int r = 0; r < 8; ++r) {
      const float v = acc12[nt][r];
      hB[w][r + 8 * g][nt * 16 + rsel] = (bf16)(v > 0.f ? v : 0.f);
    }
  WFENCE();

  // msg2 = t2 @ c_wm2 -> fB[:,0:96] f32
#pragma unroll
  for (int i = 0; i < 6; ++i) acc6[i] = ZERO8;
  gemm16<6, 6>(&hB[w][0][0], D2_F, cwm2T, D2_F, acc6);
  WFENCE();
#pragma unroll
  for (int nt = 0; nt < 6; ++nt)
#pragma unroll
    for (int r = 0; r < 8; ++r)
      fB[w][r + 8 * g][nt * 16 + rsel] = acc6[nt][r];

  wait_asynccnt0();  // shortcut tile now resident in fB[:,96:192]
  WFENCE();

  // LN(c_ln2) -> msg2 ; xc = x1 + msg2 ; LN(ln2 over 192) -> hA bf16
  if (lane < 16) {
    float s = 0.f, s2 = 0.f;
    for (int c = 0; c < D_F; ++c) { const float v = fB[w][lane][c]; s += v; s2 += v * v; }
    const float m = s * (1.f / D_F);
    const float rv = rsqrtf(s2 * (1.f / D_F) - m * m + 1e-5f);
    for (int c = 0; c < D_F; ++c) {
      const float msg2 = ((fB[w][lane][c] - m) * rv) * cln2g[c] + cln2b[c];
      fB[w][lane][c] = fA[w][lane][c] + msg2;
    }
    float t = 0.f, t2 = 0.f;
    for (int c = 0; c < D2_F; ++c) { const float v = fB[w][lane][c]; t += v; t2 += v * v; }
    const float m2 = t * (1.f / D2_F);
    const float rv2 = rsqrtf(t2 * (1.f / D2_F) - m2 * m2 + 1e-5f);
    for (int c = 0; c < D2_F; ++c)
      hA[w][lane][c] =
          (bf16)(((fB[w][lane][c] - m2) * rv2) * ln2g[c] + ln2b[c]);
  }
  WFENCE();

  // t3 = gelu(cat2 @ w_fc1 + b_fc1) -> hB bf16
#pragma unroll
  for (int i = 0; i < 12; ++i) acc12[i] = ZERO8;
  gemm16<6, 12>(&hA[w][0][0], D2_F, wfc1T, D2_F, acc12);
  WFENCE();
#pragma unroll
  for (int nt = 0; nt < 12; ++nt)
#pragma unroll
    for (int r = 0; r < 8; ++r) {
      const int col = nt * 16 + rsel;
      float v = acc12[nt][r] + bfc1[col];
      v = 0.5f * v * (1.f + erff(v * 0.70710678118f));  // exact gelu
      hB[w][r + 8 * g][col] = (bf16)v;
    }
  WFENCE();

  // out = t3 @ w_fc2 + b_fc2
#pragma unroll
  for (int i = 0; i < 6; ++i) acc6[i] = ZERO8;
  gemm16<6, 6>(&hB[w][0][0], D2_F, wfc2T, D2_F, acc6);
#pragma unroll
  for (int nt = 0; nt < 6; ++nt)
#pragma unroll
    for (int r = 0; r < 8; ++r) {
      const int col = nt * 16 + rsel;
      out[(base + r + 8 * g) * D_F + col] = acc6[nt][r] + bfc2[col];
    }
}

// ---- host -----------------------------------------------------------------
extern "C" void kernel_launch(void* const* d_in, const int* in_sizes, int n_in,
                              void* d_out, int out_size, void* d_ws,
                              size_t ws_size, hipStream_t stream) {
  const float* mr    = (const float*)d_in[0];
  // d_in[1] = ct2mr_attention_flatten : unused by the reference
  const float* warp  = (const float*)d_in[2];
  const float* ln1g  = (const float*)d_in[3];
  const float* ln1b  = (const float*)d_in[4];
  const float* wqkv  = (const float*)d_in[5];
  const float* wqkv2 = (const float*)d_in[6];
  const float* ln2g  = (const float*)d_in[7];
  const float* ln2b  = (const float*)d_in[8];
  const float* wfc1  = (const float*)d_in[9];
  const float* bfc1  = (const float*)d_in[10];
  const float* wfc2  = (const float*)d_in[11];
  const float* bfc2  = (const float*)d_in[12];
  const float* cwq   = (const float*)d_in[13];
  const float* cwk   = (const float*)d_in[14];
  const float* cwv   = (const float*)d_in[15];
  const float* cwm   = (const float*)d_in[16];
  const float* cwm1  = (const float*)d_in[17];
  const float* cwm2  = (const float*)d_in[18];
  const float* cln1g = (const float*)d_in[19];
  const float* cln1b = (const float*)d_in[20];
  const float* cln2g = (const float*)d_in[21];
  const float* cln2b = (const float*)d_in[22];
  float* out = (float*)d_out;

  char* base = (char*)d_ws;
  size_t off = 0;
  auto take = [&](size_t bytes) -> void* {
    void* p = base + off;
    off = (off + bytes + 255) & ~(size_t)255;
    return p;
  };

  float* kvb = (float*)take((size_t)(4608 + 384) * sizeof(float));
  float* ksb = kvb + 4608;
  bf16* wqkvT  = (bf16*)take((size_t)D_F * D_F * 2);
  bf16* wqkv2T = (bf16*)take((size_t)D_F * D_F * 2);
  bf16* cwqT   = (bf16*)take((size_t)D_F * D_F * 2);
  bf16* cwkT   = (bf16*)take((size_t)D_F * D_F * 2);
  bf16* cwvT   = (bf16*)take((size_t)D_F * D_F * 2);
  bf16* cwmT   = (bf16*)take((size_t)D_F * D_F * 2);
  bf16* cwm1T  = (bf16*)take((size_t)D2_F * D2_F * 2);
  bf16* cwm2T  = (bf16*)take((size_t)D2_F * D_F * 2);
  bf16* wfc1T  = (bf16*)take((size_t)D2_F * D2_F * 2);
  bf16* wfc2T  = (bf16*)take((size_t)D2_F * D_F * 2);
  float* scb = (float*)take((size_t)NL * D_F * sizeof(float));
  float* x1b = (float*)take((size_t)NL * D_F * sizeof(float));
  float* qb  = (float*)take((size_t)NL * D_F * sizeof(float));

  k0_zero<<<(4992 + 255) / 256, 256, 0, stream>>>(kvb, 4992);

  const int g96  = (D_F * D_F + 255) / 256;
  const int g192 = (D2_F * D2_F + 255) / 256;
  const int g12  = (D2_F * D_F + 255) / 256;
  k0_convT<<<g96, 256, 0, stream>>>(wqkv,  wqkvT,  D_F,  D_F);
  k0_convT<<<g96, 256, 0, stream>>>(wqkv2, wqkv2T, D_F,  D_F);
  k0_convT<<<g96, 256, 0, stream>>>(cwq,   cwqT,   D_F,  D_F);
  k0_convT<<<g96, 256, 0, stream>>>(cwk,   cwkT,   D_F,  D_F);
  k0_convT<<<g96, 256, 0, stream>>>(cwv,   cwvT,   D_F,  D_F);
  k0_convT<<<g96, 256, 0, stream>>>(cwm,   cwmT,   D_F,  D_F);
  k0_convT<<<g192, 256, 0, stream>>>(cwm1, cwm1T, D2_F, D2_F);
  k0_convT<<<g12,  256, 0, stream>>>(cwm2, cwm2T, D2_F, D_F);
  k0_convT<<<g192, 256, 0, stream>>>(wfc1, wfc1T, D2_F, D2_F);
  k0_convT<<<g12,  256, 0, stream>>>(wfc2, wfc2T, D2_F, D_F);

  k1_qkv<<<NL / 16 / 8, 256, 0, stream>>>(mr, ln1g, ln1b, wqkvT, wqkv2T, cwqT,
                                          scb, x1b, qb);
  k2_kv<<<NS / 16 / K2_TPW / 4, 128, 0, stream>>>(warp, cwkT, cwvT, kvb, ksb);
  k3_final<<<NL / 16 / 2, 64, 0, stream>>>(
      x1b, scb, qb, kvb, ksb, cwmT, cwm1T, cwm2T, wfc1T, wfc2T, cln1g, cln1b,
      cln2g, cln2b, ln2g, ln2b, bfc1, bfc2, out);
}
